// DemandMap_46755013984473
// MI455X (gfx1250) — compile-verified, matrix operand order checked by reading, not run on hardware
//
#include <hip/hip_runtime.h>
#include <stdint.h>

// ---- CDNA5 vector types -----------------------------------------------------
typedef __attribute__((ext_vector_type(16))) _Float16     v16h;
typedef __attribute__((ext_vector_type(8)))  float        v8f;
typedef __attribute__((ext_vector_type(4)))  unsigned int v4u;
typedef __attribute__((ext_vector_type(8)))  int          v8i;
typedef __attribute__((ext_vector_type(4)))  int          v4i;

#define GW    2048          // width == height == num_bins_x == num_bins_y
#define LDSW  132           // 128 tile columns + 4-column left halo

// One block = 16 rows x 128 cols of the output; 8 waves, each wave owns one
// 16x16 tile and computes it with two v_wmma_f32_16x16x32_f16 (types 2 and 3).
__global__ __launch_bounds__(256) void demand_map_kernel(
    const int* __restrict__ stm, float* __restrict__ out) {
  // 16 rows x 132 ints staged by the TDM, +64 ints slack so the (zero-weighted)
  // high-k lanes of the last wave's A fragment never read past the allocation.
  __shared__ int tile[16 * LDSW + 64];

  const int tid  = threadIdx.x;
  const int wv   = tid >> 5;
  const int lane = tid & 31;
  const int C0   = blockIdx.x * 128;  // first output column of this block
  const int R0   = blockIdx.y * 16;   // first output row of this block

  // Zero the 4-column left halo when there is no tile to the left.
  if (C0 == 0 && tid < 64) {
    tile[(tid >> 2) * LDSW + (tid & 3)] = 0;
  }
  __syncthreads();

  if (wv == 0) {
    // ---- Tensor DMA descriptor (D#): 16-row x tileW-col int32 tile --------
    const bool firstCol = (C0 == 0);
    const uint32_t tileW = firstCol ? 128u : 132u;      // tile_dim0 (elements)
    const int      gcol0 = firstCol ? 0 : (C0 - 4);
    const uint64_t gaddr = (uint64_t)(uintptr_t)stm +
                           ((uint64_t)R0 * GW + (uint64_t)gcol0) * 4u;
    const uint32_t laddr = (uint32_t)(uintptr_t)(&tile[0]) + (firstCol ? 16u : 0u);

    v4u g0;
    g0[0] = 1u;                                         // count=1, user mode
    g0[1] = laddr;                                      // lds_addr (bytes)
    g0[2] = (uint32_t)(gaddr & 0xFFFFFFFFu);            // global_addr[31:0]
    g0[3] = (uint32_t)((gaddr >> 32) & 0x01FFFFFFu)     // global_addr[56:32]
          | (2u << 30);                                 // type = 2 ("image")

    uint32_t flags = (2u << 16);                        // data_size = 4 bytes
    if (firstCol) {
      // 128-dword rows padded by 4 dwords -> LDS row stride stays 132 ints.
      flags |= (1u << 20)      // pad_enable
             | (6u << 22)      // pad_interval: 128 dwords
             | (3u << 25);     // pad_amount:   4 dwords
    }
    v8i g1;
    g1[0] = (int)flags;                                     // mask=0, flags
    g1[1] = (int)((uint32_t)(GW & 0xFFFF) << 16);           // tensor_dim0 lo16
    g1[2] = (int)(((uint32_t)GW >> 16) |
                  ((uint32_t)(GW & 0xFFFF) << 16));         // dim0 hi / dim1 lo
    g1[3] = (int)(((uint32_t)GW >> 16) | (tileW << 16));    // dim1 hi / tile_dim0
    g1[4] = 16;                                             // tile_dim1=16, tile_dim2=0
    g1[5] = GW;                                             // tensor_dim0_stride lo32
    g1[6] = 0;                                              // stride0 hi / stride1 lo
    g1[7] = 0;                                              // stride1 hi

    v4i g2 = {};                                            // unused (2D tile)
    v4i g3 = {};
    v8i g4 = {};                                            // extra group (6-arg form)

    __builtin_amdgcn_tensor_load_to_lds(g0, g1, g2, g3, g4, 0); // tensor_load_to_lds
    __builtin_amdgcn_s_wait_tensorcnt(0);                       // s_wait_tensorcnt 0
  }
  __syncthreads();

  // ---- Stencil-as-GEMM: D[m][n] = sum_k A[m][k] * B[k][n] -----------------
  // A[m][k] = indicator(stm[R0+m][Ct-4+k]==t), 16x32 f16 fragment from LDS.
  // B[k][n] = w_t[(n+4)-k], banded constant kernel (zero for k>19).
  const int n  = lane & 15;     // output column within tile / A row m
  const int hi = lane >> 4;
  const int cT = C0 + wv * 16;  // first output column of this wave's tile

  v16h a2, a3, b2, b3;
#pragma unroll
  for (int e = 0; e < 16; ++e) {
    // f16 A-fragment K index held by this lane at vector element e.
    const int k = e + 8 * hi + ((e >= 8) ? 8 : 0);
    const int s = tile[n * LDSW + wv * 16 + k];
    a2[e] = (_Float16)((s == 2) ? 1.0f : 0.0f);
    a3[e] = (_Float16)((s == 3) ? 1.0f : 0.0f);
    const int d = n + 4 - k;    // backward-stencil tap index
    b3[e] = (_Float16)((d >= 0 && d <= 4) ? 1.0f : 0.0f);
    b2[e] = (_Float16)((d == 0 || d == 1) ? 1.0f : ((d == 2) ? 0.5f : 0.0f));
  }

  v8f zero = {};
  v8f acc2 = __builtin_amdgcn_wmma_f32_16x16x32_f16(
      false, a2, false, b2, (short)0, zero, false, false);
  v8f acc3 = __builtin_amdgcn_wmma_f32_16x16x32_f16(
      false, a3, false, b3, (short)0, zero, false, false);

  // ---- Stores from WMMA D layout: VGPR v, lane -> (M = v + 8*hi, N = n) ----
  const uint32_t MAP = (uint32_t)GW * (uint32_t)GW;
#pragma unroll
  for (int vv = 0; vv < 8; ++vv) {
    const int m = vv + 8 * hi;
    const int s = tile[m * LDSW + wv * 16 + 4 + n];     // halo offset +4
    const float o0 = 1.0f - ((s == 1) ? 1.0f : 0.0f);
    const uint32_t idx =
        (uint32_t)(R0 + m) * (uint32_t)GW + (uint32_t)(cT + n);
    out[idx]           = o0;                 // binCapMap0
    out[MAP + idx]     = o0;                 // binCapMap1 (alias of map0)
    out[2u * MAP + idx] = 1.0f - acc2[vv];   // binCapMap2
    out[3u * MAP + idx] = 1.0f - acc3[vv];   // binCapMap3
  }
}

extern "C" void kernel_launch(void* const* d_in, const int* in_sizes, int n_in,
                              void* d_out, int out_size, void* d_ws, size_t ws_size,
                              hipStream_t stream) {
  const int* stm = (const int*)d_in[0];   // site_type_map, 2048*2048 int32
  float* out = (float*)d_out;             // 4 maps of 2048*2048 f32, concatenated
  dim3 grid(GW / 128, GW / 16);           // (16, 128) blocks
  demand_map_kernel<<<grid, dim3(256), 0, stream>>>(stm, out);
}
